// S4Model_Extractor_84834194030822
// MI455X (gfx1250) — matrix operand order repaired and checked by reading, not compiled, not run on hardware
//
#include <hip/hip_runtime.h>
#include <math.h>

#define B_      16
#define L_      2048
#define DIN     64
#define H_      256
#define NH      32
#define NLAYERS 4

typedef __attribute__((ext_vector_type(16))) _Float16 v16h;
typedef __attribute__((ext_vector_type(8)))  float    v8f;

// ---------------------------------------------------------------------------
// Encoder: h[b,c,l] = x[b,l,:] @ enc_w[:,c] + enc_b[c]
// GEMM (32768 x 64) @ (64 x 256) via v_wmma_f32_16x16x32_f16, K in 2 chunks.
// Output written transposed into (B,H,L) layout for the scan.
// ---------------------------------------------------------------------------
__global__ void __launch_bounds__(256) enc_wmma_kernel(
    const float* __restrict__ x, const float* __restrict__ w,
    const float* __restrict__ bias, float* __restrict__ h) {
  const int lane = threadIdx.x & 31;
  const int wid  = (blockIdx.x << 3) | (threadIdx.x >> 5); // tile id [0,32768)
  const int mt   = wid >> 4;     // 2048 row tiles (rows = b*L+l)
  const int nt   = wid & 15;     // 16 col tiles (H=256)
  const int half = lane >> 4;
  const int rlo  = lane & 15;
  const int R    = mt * 16 + rlo;      // A row held by this lane
  const int col  = nt * 16 + rlo;      // B column held by this lane
  v8f acc = {};
#pragma unroll
  for (int kc = 0; kc < DIN; kc += 32) {
    const float* xr = x + (size_t)R * DIN;
    v16h a, bm;
#pragma unroll
    for (int e = 0; e < 8; ++e) {
      a[e]     = (_Float16)xr[kc + half * 16 + e];       // K = kc + {0..7|16..23}
      a[8 + e] = (_Float16)xr[kc + half * 16 + 8 + e];   // K = kc + {8..15|24..31}
    }
    const int kb = kc + half * 16;                       // B: K 0-15 / 16-31 per half
#pragma unroll
    for (int e = 0; e < 16; ++e)
      bm[e] = (_Float16)w[(size_t)(kb + e) * H_ + col];
    acc = __builtin_amdgcn_wmma_f32_16x16x32_f16(false, a, false, bm,
                                                 (short)0, acc, false, false);
  }
  const float bc = bias[col];
#pragma unroll
  for (int r = 0; r < 8; ++r) {
    int m  = r + half * 8;               // D row of this vgpr
    int Rm = mt * 16 + m;
    int bb = Rm >> 11;                   // / L
    int l  = Rm & (L_ - 1);
    h[((size_t)bb * H_ + col) * L_ + l] = acc[r] + bc;
  }
}

// ---------------------------------------------------------------------------
// S4D layer core: exact diagonal-SSM recurrence == reference FFT convolution.
// One wave per (b, channel); lane = complex state n (Nh == 32 == wave32).
// Per 32-timestep chunk: lane n runs the recurrence and stores c_n[t] into a
// padded LDS tile; after an explicit s_wait_dscnt, lane t reads the transposed
// column, reduces over states, and applies GELU for *its own* timestep only
// (erf amortized 32x vs per-step butterfly). Same-wave LDS is in-order, each
// wave owns a private LDS region -> no block barrier needed.
// ---------------------------------------------------------------------------
__global__ void __launch_bounds__(256) scan_gelu_kernel(
    const float* __restrict__ h, float* __restrict__ yg,
    const float* __restrict__ log_dt, const float* __restrict__ log_A_real,
    const float* __restrict__ A_imag, const float* __restrict__ C_re,
    const float* __restrict__ C_im, const float* __restrict__ D_skip) {
  __shared__ float cbuf[8][NH * 33];                        // 33-stride: no bank conflicts
  const int lane = threadIdx.x & 31;                        // state index n
  const int wv   = threadIdx.x >> 5;
  const int wid  = (blockIdx.x << 3) | wv;                  // (b,hc) in [0,4096)
  const int b  = wid >> 8;
  const int hc = wid & 255;
  const int pn = hc * NH + lane;
  float* cw = &cbuf[wv][0];

  // Discretize: r = exp(dt*A), Ck = C*(r-1)/A   (complex, per state)
  const float dt  = expf(log_dt[hc]);
  const float Are = -expf(log_A_real[pn]);
  const float Aim = A_imag[pn];
  const float dre = dt * Are, dim = dt * Aim;
  const float er  = expf(dre);
  const float rre = er * cosf(dim);
  const float rim = er * sinf(dim);
  const float Ere = rre - 1.f, Eim = rim;
  const float cre = C_re[pn], cim = C_im[pn];
  const float tre = cre * Ere - cim * Eim;
  const float tim = cre * Eim + cim * Ere;
  const float den = Are * Are + Aim * Aim;
  const float ckre = (tre * Are + tim * Aim) / den;   // (C*E) * conj(A) / |A|^2
  const float ckim = (tim * Are - tre * Aim) / den;
  const float dsk  = D_skip[hc];

  const size_t base = ((size_t)b * H_ + hc) * L_;
  float xre = 0.f, xim = 0.f;
  for (int t0 = 0; t0 < L_; t0 += 32) {
    const float uv = h[base + t0 + lane];   // 32 inputs, one per lane (coalesced)
    // recurrence: lane = state, 32 steps; stage contributions to LDS
#pragma unroll
    for (int tt = 0; tt < 32; ++tt) {
      float u   = __shfl(uv, tt, 32);       // broadcast u[t0+tt]
      float nre = rre * xre - rim * xim + u;
      float nim = rre * xim + rim * xre;
      xre = nre; xim = nim;
      cw[lane * 33 + tt] = ckre * xre - ckim * xim;   // ds_store_b32
    }
    asm volatile("s_wait_dscnt 0x0" ::: "memory");    // stores visible to loads
    // transpose-reduce: lane t sums over all 32 states for its own timestep
    float s = 0.f;
#pragma unroll
    for (int n = 0; n < NH; ++n) s += cw[n * 33 + lane];
    float y = 2.f * s + dsk * uv;                     // uv is u at this lane's t
    float g = 0.5f * y * (1.f + erff(y * 0.70710678118654752f)); // exact gelu
    yg[base + t0 + lane] = g;                         // coalesced store
  }
}

// ---------------------------------------------------------------------------
// output_linear + GLU + residual (fused, in place):
//   h[b,o,l] += a*sigmoid(g), where [a;g] = w_out @ yg + b_out.
// Each wave owns one o-tile pair x FOUR l-tiles: the A/G fragments of w_out
// are loaded once per K-chunk and amortized over 4 B-tiles (A traffic / 4).
// ---------------------------------------------------------------------------
#define NB 4
__global__ void __launch_bounds__(256) glu_wmma_kernel(
    const float* __restrict__ yg, const float* __restrict__ wout,
    const float* __restrict__ bout, float* __restrict__ h) {
  const int lane = threadIdx.x & 31;
  const int wid  = (blockIdx.x << 3) | (threadIdx.x >> 5); // [0, 8192)
  const int b    = wid >> 9;           // 16 batches
  const int rem  = wid & 511;
  const int ot   = rem >> 5;           // 16 o-tiles (rows of the 'a' half)
  const int ltg  = rem & 31;           // 32 groups of 4 l-tiles
  const int half = lane >> 4;
  const int rlo  = lane & 15;
  const int oa   = ot * 16 + rlo;      // 'a' row
  const int og   = oa + H_;            // 'g' row
  const int l0   = ltg * (16 * NB) + rlo;
  const float* wa  = wout + (size_t)oa * H_;
  const float* wg  = wout + (size_t)og * H_;
  const float* ygb = yg + (size_t)b * H_ * L_;
  v8f acca[NB], accg[NB];
#pragma unroll
  for (int j = 0; j < NB; ++j) { acca[j] = (v8f){}; accg[j] = (v8f){}; }

  for (int kc = 0; kc < H_; kc += 32) {
    v16h a, g;
#pragma unroll
    for (int e = 0; e < 8; ++e) {
      a[e]     = (_Float16)wa[kc + half * 16 + e];
      a[8 + e] = (_Float16)wa[kc + half * 16 + 8 + e];
      g[e]     = (_Float16)wg[kc + half * 16 + e];
      g[8 + e] = (_Float16)wg[kc + half * 16 + 8 + e];
    }
    const int kb = kc + half * 16;
#pragma unroll
    for (int j = 0; j < NB; ++j) {
      v16h bm;
#pragma unroll
      for (int e = 0; e < 16; ++e)
        bm[e] = (_Float16)ygb[(size_t)(kb + e) * L_ + l0 + j * 16];
      acca[j] = __builtin_amdgcn_wmma_f32_16x16x32_f16(false, a, false, bm,
                                                       (short)0, acca[j], false, false);
      accg[j] = __builtin_amdgcn_wmma_f32_16x16x32_f16(false, g, false, bm,
                                                       (short)0, accg[j], false, false);
    }
  }
#pragma unroll
  for (int j = 0; j < NB; ++j) {
#pragma unroll
    for (int r = 0; r < 8; ++r) {
      int m = r + half * 8;
      int o = ot * 16 + m;
      float av = acca[j][r] + bout[o];
      float gv = accg[j][r] + bout[o + H_];
      float z  = av / (1.f + __expf(-gv));          // a * sigmoid(g)
      h[((size_t)b * H_ + o) * L_ + l0 + j * 16] += z;  // residual, in place
    }
  }
}

// ---------------------------------------------------------------------------
// Post-LayerNorm over channels, in place. Thread per (b,l); lanes hit
// consecutive l -> every channel pass is coalesced (stride-L per step).
// ---------------------------------------------------------------------------
__global__ void __launch_bounds__(256) ln_kernel(
    float* __restrict__ h, const float* __restrict__ g,
    const float* __restrict__ bta) {
  const int idx = blockIdx.x * 256 + threadIdx.x;  // (b,l) in [0,32768)
  const int b = idx >> 11;
  const int l = idx & 2047;
  float* hp = h + (size_t)b * H_ * L_ + l;
  float s = 0.f, ss = 0.f;
#pragma unroll 4
  for (int c = 0; c < H_; ++c) {
    float v = hp[(size_t)c * L_];
    s += v; ss += v * v;
  }
  const float mu  = s * (1.f / H_);
  const float var = ss * (1.f / H_) - mu * mu;     // biased, as torch/jax
  const float inv = rsqrtf(var + 1e-5f);
#pragma unroll 4
  for (int c = 0; c < H_; ++c) {
    float v = hp[(size_t)c * L_];
    hp[(size_t)c * L_] = (v - mu) * inv * g[c] + bta[c];
  }
}

// Mean pool over L: one wave per (b,c).
__global__ void __launch_bounds__(256) pool_kernel(
    const float* __restrict__ h, float* __restrict__ pooled) {
  const int lane = threadIdx.x & 31;
  const int wid  = (blockIdx.x << 3) | (threadIdx.x >> 5); // (b,c) in [0,4096)
  const float* hp = h + (size_t)wid * L_;
  float s = 0.f;
  for (int t = lane; t < L_; t += 32) s += hp[t];
#pragma unroll
  for (int off = 16; off > 0; off >>= 1) s += __shfl_xor(s, off, 32);
  if (lane == 0) pooled[wid] = s * (1.f / L_);
}

// Tiny decoder MLP (16x256 -> 128 -> 64 -> 32) in one block via LDS staging.
__global__ void __launch_bounds__(256) dec_kernel(
    const float* __restrict__ pooled,
    const float* __restrict__ w1, const float* __restrict__ b1,
    const float* __restrict__ w2, const float* __restrict__ b2,
    const float* __restrict__ w3, const float* __restrict__ b3,
    float* __restrict__ out) {
  __shared__ float sp[B_ * H_];
  __shared__ float s1[B_ * 128];
  __shared__ float s2[B_ * 64];
  const int tid = threadIdx.x;
  for (int i = tid; i < B_ * H_; i += 256) sp[i] = pooled[i];
  __syncthreads();
  for (int i = tid; i < B_ * 128; i += 256) {
    int r = i >> 7, j = i & 127;
    float acc = b1[j];
    for (int k = 0; k < H_; ++k) acc += sp[r * H_ + k] * w1[k * 128 + j];
    s1[i] = fmaxf(acc, 0.f);
  }
  __syncthreads();
  for (int i = tid; i < B_ * 64; i += 256) {
    int r = i >> 6, j = i & 63;
    float acc = b2[j];
    for (int k = 0; k < 128; ++k) acc += s1[r * 128 + k] * w2[k * 64 + j];
    s2[i] = fmaxf(acc, 0.f);
  }
  __syncthreads();
  for (int i = tid; i < B_ * 32; i += 256) {
    int r = i >> 5, j = i & 31;
    float acc = b3[j];
    for (int k = 0; k < 64; ++k) acc += s2[r * 64 + k] * w3[k * 32 + j];
    out[i] = acc;
  }
}

extern "C" void kernel_launch(void* const* d_in, const int* in_sizes, int n_in,
                              void* d_out, int out_size, void* d_ws, size_t ws_size,
                              hipStream_t stream) {
  const float* x          = (const float*)d_in[0];
  const float* enc_w      = (const float*)d_in[1];
  const float* enc_b      = (const float*)d_in[2];
  const float* log_dt     = (const float*)d_in[3];
  const float* C_re       = (const float*)d_in[4];
  const float* C_im       = (const float*)d_in[5];
  const float* log_A_real = (const float*)d_in[6];
  const float* A_imag     = (const float*)d_in[7];
  const float* D_skip     = (const float*)d_in[8];
  const float* w_out      = (const float*)d_in[9];
  const float* b_out      = (const float*)d_in[10];
  const float* ln_g       = (const float*)d_in[11];
  const float* ln_b       = (const float*)d_in[12];
  const float* w1 = (const float*)d_in[13];
  const float* b1 = (const float*)d_in[14];
  const float* w2 = (const float*)d_in[15];
  const float* b2 = (const float*)d_in[16];
  const float* w3 = (const float*)d_in[17];
  const float* b3 = (const float*)d_in[18];

  // Workspace: h (33.5 MB) | yg (33.5 MB) | pooled (16 KB)  => ~67.1 MB total
  float* h      = (float*)d_ws;
  float* yg     = h + (size_t)B_ * H_ * L_;
  float* pooled = yg + (size_t)B_ * H_ * L_;

  enc_wmma_kernel<<<4096, 256, 0, stream>>>(x, enc_w, enc_b, h);
  for (int i = 0; i < NLAYERS; ++i) {
    scan_gelu_kernel<<<512, 256, 0, stream>>>(
        h, yg,
        log_dt + (size_t)i * H_,
        log_A_real + (size_t)i * H_ * NH,
        A_imag + (size_t)i * H_ * NH,
        C_re + (size_t)i * H_ * NH,
        C_im + (size_t)i * H_ * NH,
        D_skip + (size_t)i * H_);
    glu_wmma_kernel<<<1024, 256, 0, stream>>>(
        yg, w_out + (size_t)i * 2 * H_ * H_, b_out + (size_t)i * 2 * H_, h);
    ln_kernel<<<128, 256, 0, stream>>>(h, ln_g + (size_t)i * H_,
                                       ln_b + (size_t)i * H_);
  }
  pool_kernel<<<512, 256, 0, stream>>>(h, pooled);
  dec_kernel<<<1, 256, 0, stream>>>(pooled, w1, b1, w2, b2, w3, b3,
                                    (float*)d_out);
}
